// Model_71768903516445
// MI455X (gfx1250) — compile-verified, matrix operand order checked by reading, not run on hardware
//
#include <hip/hip_runtime.h>
#include <hip/hip_bf16.h>
#include <stdint.h>

// ---------- Model dims ----------
#define VOCAB 16000
#define EMBED 16
#define EPAD  32      // K padded to WMMA granule
#define HID   256
#define BATCH 64
#define TSTEPS 128
#define G4H   1024    // 4*HID

typedef __attribute__((ext_vector_type(16))) __bf16 v16bf;
typedef __attribute__((ext_vector_type(8)))  float  v8f;
typedef __attribute__((ext_vector_type(4)))  unsigned int u32x4;

// explicit global-address-space pointers (keep loads as global_load_*, not flat_*)
typedef const __attribute__((address_space(1))) unsigned short* gwp;
typedef __attribute__((address_space(1))) unsigned short* gwp_mut;
typedef const __attribute__((address_space(1))) u32x4* gq4p;

__device__ inline unsigned short f2bf(float f) {
  unsigned int x = __float_as_uint(f);
  unsigned int r = x + 0x7FFFu + ((x >> 16) & 1u);
  return (unsigned short)(r >> 16);
}

__device__ inline float sigmoidf_(float x) { return 1.0f / (1.0f + __expf(-x)); }

__device__ inline v8f vzero8() {
  v8f z = {0.f, 0.f, 0.f, 0.f, 0.f, 0.f, 0.f, 0.f};
  return z;
}

// Load one 16x32 bf16 operand tile per the CDNA5 16-bit matrix lane layout:
// lanes 0-15 take K=0..7 (VGPR0-3) and K=16..23 (VGPR4-7) contiguously;
// lanes 16-31 take K=8..15 and K=24..31.  -> two 16-byte loads per lane.
__device__ inline v16bf load_tile(const unsigned short* p, int lane) {
  union U { unsigned int u[8]; v16bf v; } t;
  const unsigned short* q = p + ((lane & 16) ? 8 : 0);
  const uint4 lo = *(const uint4*)(q);
  const uint4 hi = *(const uint4*)(q + 16);
  t.u[0] = lo.x; t.u[1] = lo.y; t.u[2] = lo.z; t.u[3] = lo.w;
  t.u[4] = hi.x; t.u[5] = hi.y; t.u[6] = hi.z; t.u[7] = hi.w;
  return t.v;
}

// Same, but through an explicit global (AS1) pointer -> global_load_b128.
__device__ inline v16bf load_tile_g(gwp p, int lane) {
  union U { unsigned int u[8]; v16bf v; } t;
  gwp q = p + ((lane & 16) ? 8 : 0);
  u32x4 lo = *(gq4p)(q);
  u32x4 hi = *(gq4p)(q + 16);
  t.u[0] = lo[0]; t.u[1] = lo[1]; t.u[2] = lo[2]; t.u[3] = lo[3];
  t.u[4] = hi[0]; t.u[5] = hi[1]; t.u[6] = hi[2]; t.u[7] = hi[3];
  return t.v;
}

// ---------- Prep kernels ----------
__global__ void conv_bf16_pad(const float* __restrict__ src,
                              unsigned short* __restrict__ dst,
                              int rows, int sc, int dc) {
  int i = blockIdx.x * blockDim.x + threadIdx.x;
  if (i >= rows * dc) return;
  int r = i / dc, c = i % dc;
  dst[i] = (c < sc) ? f2bf(src[r * sc + c]) : (unsigned short)0;
}

__global__ void vec_add_kernel(const float* __restrict__ a,
                               const float* __restrict__ b,
                               float* __restrict__ o, int n) {
  int i = blockIdx.x * blockDim.x + threadIdx.x;
  if (i < n) o[i] = a[i] + b[i];
}

__global__ void embed_kernel(const int* __restrict__ tok,
                             const float* __restrict__ emb,
                             unsigned short* __restrict__ xe) {
  int i = blockIdx.x * blockDim.x + threadIdx.x;
  if (i >= BATCH * TSTEPS * EPAD) return;
  int e = i & (EPAD - 1);
  int bt = i >> 5;           // b*T + t
  int b = bt >> 7;
  int t = bt & (TSTEPS - 1);
  int token = tok[b * TSTEPS + t];
  unsigned short v = 0;
  if (e < EMBED) v = f2bf(emb[token * EMBED + e]);
  xe[(t * BATCH + b) * EPAD + e] = v;   // [T][B][EPAD]
}

__global__ void labels_kernel(const int* __restrict__ tok, float* __restrict__ lab) {
  int i = blockIdx.x * blockDim.x + threadIdx.x;
  if (i < BATCH * TSTEPS) lab[i] = (float)tok[i];
}

// ---------- Phase 2: sequential 3-layer LSTM scan + head-L1, one WGP ----------
// 32 waves; the [64,256] activation grid is 4(M) x 16(N) = 64 16x16 tiles;
// each wave owns 2 tiles for the entire run, so cell state c lives in VGPRs.
// h state is DOUBLE-BUFFERED in LDS (192 KB): each layer reads the old buffer
// and writes the new one -> one barrier per layer, one position in flight.
// Weight tiles are re-fetched from global (L2-resident, shared across waves)
// every iteration via AS1 pointers (global_load_b128, LOADcnt only); pointer
// laundering stops LICM from hoisting + spilling them.
__device__ inline void compute_gates(const unsigned short* Ax, int ldx, int kx,
                                     gwp Wih, int ldw,
                                     const unsigned short* Ah,
                                     gwp Whh,
                                     int m0, int n0, int lane, v8f acc[4]) {
  const unsigned short* axr = Ax + (m0 + (lane & 15)) * ldx;
  const unsigned short* ahr = Ah + (m0 + (lane & 15)) * HID;
  const int nrow = n0 + (lane & 15);
#pragma unroll
  for (int g = 0; g < 4; ++g) acc[g] = vzero8();
  // x-part: one A tile feeds all four gates' WMMAs
  for (int k = 0; k < kx; k += 32) {
    v16bf a = load_tile(axr + k, lane);
#pragma unroll
    for (int g = 0; g < 4; ++g) {
      v16bf b = load_tile_g(Wih + (g * HID + nrow) * ldw + k, lane);
      acc[g] = __builtin_amdgcn_wmma_f32_16x16x32_bf16(false, a, false, b, (short)0, acc[g], false, false);
    }
  }
  // h-part
  for (int k = 0; k < HID; k += 32) {
    v16bf a = load_tile(ahr + k, lane);
#pragma unroll
    for (int g = 0; g < 4; ++g) {
      v16bf b = load_tile_g(Whh + (g * HID + nrow) * HID + k, lane);
      acc[g] = __builtin_amdgcn_wmma_f32_16x16x32_bf16(false, a, false, b, (short)0, acc[g], false, false);
    }
  }
}

__global__ __launch_bounds__(1024) void lstm_scan_kernel(
    const unsigned short* __restrict__ xe,                       // [T][64][32] bf16
    const unsigned short* __restrict__ WIH1p,                    // [1024][32]
    const unsigned short* __restrict__ WHH1p,                    // [1024][256]
    const float* __restrict__ bc1,
    const unsigned short* __restrict__ WIH2p,
    const unsigned short* __restrict__ WHH2p,
    const float* __restrict__ bc2,
    const unsigned short* __restrict__ WIH3p,
    const unsigned short* __restrict__ WHH3p,
    const float* __restrict__ bc3,
    const unsigned short* __restrict__ W1Bp,                     // [256][256]
    const float* __restrict__ b1,
    const float* __restrict__ h0, const float* __restrict__ c0,
    unsigned short* __restrict__ Sbuf)                           // [8192][256] bf16
{
  __shared__ unsigned short sh_h[2][3][BATCH][HID];              // 192 KB

  const int tid  = threadIdx.x;
  const int lane = tid & 31;
  const int wave = tid >> 5;

  // launderable, explicit-global copies of the weight bases
  gwp wih1 = (gwp)WIH1p;
  gwp whh1 = (gwp)WHH1p;
  gwp wih2 = (gwp)WIH2p;
  gwp whh2 = (gwp)WHH2p;
  gwp wih3 = (gwp)WIH3p;
  gwp whh3 = (gwp)WHH3p;
  gwp w1b  = (gwp)W1Bp;
  gwp_mut sb = (gwp_mut)Sbuf;

  // init h buffer 0 (same h0 for all 3 layers)
  for (int i = tid; i < BATCH * HID; i += 1024) {
    unsigned short v = f2bf(h0[i]);
    sh_h[0][0][0][i] = v;
    sh_h[0][1][0][i] = v;
    sh_h[0][2][0][i] = v;
  }

  // tile ownership: pos = wave*2 + p ; pos -> (mt, nt)
  int m0a[2], n0a[2];
#pragma unroll
  for (int p = 0; p < 2; ++p) {
    int pos = wave * 2 + p;
    m0a[p] = (pos >> 4) * 16;
    n0a[p] = (pos & 15) * 16;
  }

  // cell states in registers (same c0 for all 3 layers)
  float c1r[2][8], c2r[2][8], c3r[2][8];
#pragma unroll
  for (int p = 0; p < 2; ++p) {
    int n = n0a[p] + (lane & 15);
#pragma unroll
    for (int r = 0; r < 8; ++r) {
      int m = m0a[p] + r + ((lane & 16) ? 8 : 0);
      float cv = c0[m * HID + n];
      c1r[p][r] = cv; c2r[p][r] = cv; c3r[p][r] = cv;
    }
  }
  __syncthreads();

  auto run_layer = [&](const unsigned short* Ax, int ldx, int kx,
                       gwp Wih, int ldw,
                       gwp Whh, const float* bc,
                       const unsigned short* hOld, unsigned short* hNew,
                       float (&cL)[2][8]) {
#pragma unroll
    for (int p = 0; p < 2; ++p) {
      v8f acc[4];
      compute_gates(Ax, ldx, kx, Wih, ldw, hOld, Whh, m0a[p], n0a[p], lane, acc);
      int n = n0a[p] + (lane & 15);
      float bi = bc[0 * HID + n], bf = bc[1 * HID + n];
      float bg = bc[2 * HID + n], bo = bc[3 * HID + n];
#pragma unroll
      for (int r = 0; r < 8; ++r) {
        int m = m0a[p] + r + ((lane & 16) ? 8 : 0);
        float iv = sigmoidf_(acc[0][r] + bi);
        float fv = sigmoidf_(acc[1][r] + bf);
        float gv = tanhf(acc[2][r] + bg);
        float ov = sigmoidf_(acc[3][r] + bo);
        float cn = fv * cL[p][r] + iv * gv;
        cL[p][r] = cn;
        hNew[m * HID + n] = f2bf(ov * tanhf(cn));
      }
    }
  };

#pragma unroll 1
  for (int t = 0; t < TSTEPS; ++t) {
    // Opaque-ify the weight bases each iteration so the weight-tile loads
    // cannot be hoisted out of the time loop (which would spill ~70 tiles
    // to per-wave scratch).  AS1 type keeps them as global_load_b128.
    asm volatile("" : "+s"(wih1), "+s"(whh1), "+s"(wih2), "+s"(whh2),
                      "+s"(wih3), "+s"(whh3), "+s"(w1b));

    const int cur = t & 1, nxt = cur ^ 1;
    const unsigned short* xt = xe + t * BATCH * EPAD;
    unsigned short*       h1n = &sh_h[nxt][0][0][0];
    unsigned short*       h2n = &sh_h[nxt][1][0][0];
    unsigned short*       h3n = &sh_h[nxt][2][0][0];
    const unsigned short* h1o = &sh_h[cur][0][0][0];
    const unsigned short* h2o = &sh_h[cur][1][0][0];
    const unsigned short* h3o = &sh_h[cur][2][0][0];

    run_layer(xt,  EPAD, EPAD, wih1, EPAD, whh1, bc1, h1o, h1n, c1r);
    __syncthreads();                       // h1n complete
    run_layer(h1n, HID,  HID,  wih2, HID,  whh2, bc2, h2o, h2n, c2r);
    __syncthreads();                       // h2n complete
    run_layer(h2n, HID,  HID,  wih3, HID,  whh3, bc3, h3o, h3n, c3r);
    __syncthreads();                       // h3n complete

    // head layer 1: S[b*T+t, :] = relu(h3 @ W1^T + b1), stored bf16.
    // Reads h3n, which is not overwritten until t+2 -> no trailing barrier.
#pragma unroll
    for (int p = 0; p < 2; ++p) {
      v8f acc = vzero8();
      const unsigned short* ar = h3n + (m0a[p] + (lane & 15)) * HID;
      gwp wr = w1b + (n0a[p] + (lane & 15)) * HID;
      for (int k = 0; k < HID; k += 32) {
        v16bf a = load_tile(ar + k, lane);
        v16bf b = load_tile_g(wr + k, lane);
        acc = __builtin_amdgcn_wmma_f32_16x16x32_bf16(false, a, false, b, (short)0, acc, false, false);
      }
      int n = n0a[p] + (lane & 15);
      float bb = b1[n];
#pragma unroll
      for (int r = 0; r < 8; ++r) {
        int m = m0a[p] + r + ((lane & 16) ? 8 : 0);
        float v = acc[r] + bb;
        v = v > 0.f ? v : 0.f;
        sb[(m * TSTEPS + t) * HID + n] = f2bf(v);
      }
    }
  }
}

// ---------- Phase 3: head GEMM  out[8192,16000] = S @ W2b^T + b2 ----------
// One wave computes a 16(M) x 64(N) strip: 4 accumulators share each A tile.
__global__ __launch_bounds__(256) void head_gemm_kernel(
    const unsigned short* __restrict__ Sp,    // [8192][256] bf16
    const unsigned short* __restrict__ W2Bp,  // [16000][256] bf16
    const float* __restrict__ b2,
    float* __restrict__ out)                  // [8192][16000] f32
{
  const int wid  = blockIdx.x * 8 + (threadIdx.x >> 5);
  const int lane = threadIdx.x & 31;
  const int mt = wid / (VOCAB / 64);          // 0..511
  const int ng = wid % (VOCAB / 64);          // 0..249
  if (mt >= (BATCH * TSTEPS) / 16) return;    // wave-uniform guard
  const int m0 = mt * 16;
  const int n0 = ng * 64;

  gwp S   = (gwp)Sp;
  gwp W2B = (gwp)W2Bp;

  v8f acc[4] = {vzero8(), vzero8(), vzero8(), vzero8()};
  gwp ar = S + (m0 + (lane & 15)) * HID;
  for (int k = 0; k < HID; k += 32) {
    v16bf a = load_tile_g(ar + k, lane);
#pragma unroll
    for (int j = 0; j < 4; ++j) {
      gwp br = W2B + (n0 + j * 16 + (lane & 15)) * HID;
      v16bf b = load_tile_g(br + k, lane);
      acc[j] = __builtin_amdgcn_wmma_f32_16x16x32_bf16(false, a, false, b, (short)0, acc[j], false, false);
    }
  }
#pragma unroll
  for (int j = 0; j < 4; ++j) {
    int n = n0 + j * 16 + (lane & 15);
    float bb = b2[n];
#pragma unroll
    for (int r = 0; r < 8; ++r) {
      int m = m0 + r + ((lane & 16) ? 8 : 0);
      out[(size_t)m * VOCAB + n] = acc[j][r] + bb;
    }
  }
}

// ---------- Host ----------
extern "C" void kernel_launch(void* const* d_in, const int* in_sizes, int n_in,
                              void* d_out, int out_size, void* d_ws, size_t ws_size,
                              hipStream_t stream) {
  const int*   tokens = (const int*)  d_in[0];
  const float* h0     = (const float*)d_in[1];
  const float* c0     = (const float*)d_in[2];
  const float* emb    = (const float*)d_in[3];
  const float* Wih1   = (const float*)d_in[4];
  const float* Whh1   = (const float*)d_in[5];
  const float* bih1   = (const float*)d_in[6];
  const float* bhh1   = (const float*)d_in[7];
  const float* Wih2   = (const float*)d_in[8];
  const float* Whh2   = (const float*)d_in[9];
  const float* bih2   = (const float*)d_in[10];
  const float* bhh2   = (const float*)d_in[11];
  const float* Wih3   = (const float*)d_in[12];
  const float* Whh3   = (const float*)d_in[13];
  const float* bih3   = (const float*)d_in[14];
  const float* bhh3   = (const float*)d_in[15];
  const float* W1     = (const float*)d_in[16];
  const float* b1     = (const float*)d_in[17];
  const float* W2     = (const float*)d_in[18];
  const float* b2     = (const float*)d_in[19];

  float* out = (float*)d_out;
  float* labels = out + (size_t)BATCH * TSTEPS * VOCAB;

  char* w = (char*)d_ws;
  auto take = [&](size_t bytes) {
    char* p = w;
    w += (bytes + 255) & ~(size_t)255;
    return p;
  };
  unsigned short* WIH1 = (unsigned short*)take((size_t)G4H * EPAD * 2);
  unsigned short* WHH1 = (unsigned short*)take((size_t)G4H * HID * 2);
  unsigned short* WIH2 = (unsigned short*)take((size_t)G4H * HID * 2);
  unsigned short* WHH2 = (unsigned short*)take((size_t)G4H * HID * 2);
  unsigned short* WIH3 = (unsigned short*)take((size_t)G4H * HID * 2);
  unsigned short* WHH3 = (unsigned short*)take((size_t)G4H * HID * 2);
  unsigned short* W1B  = (unsigned short*)take((size_t)HID * HID * 2);
  unsigned short* W2B  = (unsigned short*)take((size_t)VOCAB * HID * 2);
  unsigned short* XE   = (unsigned short*)take((size_t)TSTEPS * BATCH * EPAD * 2);
  unsigned short* SB   = (unsigned short*)take((size_t)BATCH * TSTEPS * HID * 2);
  float* BC1 = (float*)take(G4H * 4);
  float* BC2 = (float*)take(G4H * 4);
  float* BC3 = (float*)take(G4H * 4);

  const int TPB = 256;
  auto blocks = [](size_t n, int t) { return (unsigned)((n + t - 1) / t); };

  // Phase 1: precision/layout prep
  conv_bf16_pad<<<blocks((size_t)G4H * EPAD, TPB), TPB, 0, stream>>>(Wih1, WIH1, G4H, EMBED, EPAD);
  conv_bf16_pad<<<blocks((size_t)G4H * HID, TPB), TPB, 0, stream>>>(Whh1, WHH1, G4H, HID, HID);
  conv_bf16_pad<<<blocks((size_t)G4H * HID, TPB), TPB, 0, stream>>>(Wih2, WIH2, G4H, HID, HID);
  conv_bf16_pad<<<blocks((size_t)G4H * HID, TPB), TPB, 0, stream>>>(Whh2, WHH2, G4H, HID, HID);
  conv_bf16_pad<<<blocks((size_t)G4H * HID, TPB), TPB, 0, stream>>>(Wih3, WIH3, G4H, HID, HID);
  conv_bf16_pad<<<blocks((size_t)G4H * HID, TPB), TPB, 0, stream>>>(Whh3, WHH3, G4H, HID, HID);
  conv_bf16_pad<<<blocks((size_t)HID * HID, TPB), TPB, 0, stream>>>(W1, W1B, HID, HID, HID);
  conv_bf16_pad<<<blocks((size_t)VOCAB * HID, TPB), TPB, 0, stream>>>(W2, W2B, VOCAB, HID, HID);
  vec_add_kernel<<<blocks(G4H, TPB), TPB, 0, stream>>>(bih1, bhh1, BC1, G4H);
  vec_add_kernel<<<blocks(G4H, TPB), TPB, 0, stream>>>(bih2, bhh2, BC2, G4H);
  vec_add_kernel<<<blocks(G4H, TPB), TPB, 0, stream>>>(bih3, bhh3, BC3, G4H);
  embed_kernel<<<blocks((size_t)BATCH * TSTEPS * EPAD, TPB), TPB, 0, stream>>>(tokens, emb, XE);
  labels_kernel<<<blocks((size_t)BATCH * TSTEPS, TPB), TPB, 0, stream>>>(tokens, labels);

  // Phase 2: sequential recurrence on one WGP (WMMA bf16, double-buffered h in LDS)
  lstm_scan_kernel<<<1, 1024, 0, stream>>>(XE,
                                           WIH1, WHH1, BC1,
                                           WIH2, WHH2, BC2,
                                           WIH3, WHH3, BC3,
                                           W1B, b1, h0, c0, SB);

  // Phase 3: bandwidth-bound head GEMM (524 MB of f32 scores)
  const size_t nwaves = ((size_t)BATCH * TSTEPS / 16) * (VOCAB / 64);  // 128000
  head_gemm_kernel<<<(unsigned)(nwaves / 8), 256, 0, stream>>>(SB, W2B, b2, out);
}
